// GATLayer_28647431864951
// MI455X (gfx1250) — matrix operand order repaired and checked by reading, not compiled, not run on hardware
//
#include <hip/hip_runtime.h>
#include <math.h>

// GATConv forward for MI455X (gfx1250, wave32).
// Pipeline (all on `stream`, in order):
//   0) gat_pack_w      : one-shot repack of W (128x128) into split-bf16 hi/lo
//                        B-fragments in WMMA lane layout (contiguous per lane).
//   1) gat_gemm_wmma   : h = x @ W via v_wmma_f32_16x16x32_bf16, split-bf16
//                        (hi+lo) for ~fp32 accuracy; 12 WMMAs per 16x16 tile.
//   2) gat_prep        : a_src/a_dst dot products, init segment-max (-inf,
//                        ordered uint encoding), segment-sum = 0, out = bias.
//   3) gat_edge_max    : atomicMax of leaky-relu logits per (dst, head).
//   4) gat_edge_sum    : atomicAdd of exp(e - m) per (dst, head).
//   5) gat_edge_scatter: wave-per-edge, out[dst] += alpha * h[src] (128 f32
//                        atomics, L2-resident).

#define F_DIM 128
#define H_HEADS 4
#define C_CH 32
#define NEG_SLOPE 0.2f
#define EPS_F 1e-16f

typedef __attribute__((ext_vector_type(16))) __bf16        v16bf;
typedef __attribute__((ext_vector_type(8)))  float         v8f;
typedef __attribute__((ext_vector_type(8)))  unsigned int  u8v;

// Order-preserving float <-> uint encoding (atomicMax over signed floats).
__device__ __forceinline__ unsigned f2ord(float f) {
  unsigned u = __float_as_uint(f);
  return (u & 0x80000000u) ? ~u : (u | 0x80000000u);
}
__device__ __forceinline__ float ord2f(unsigned u) {
  return __uint_as_float((u & 0x80000000u) ? (u & 0x7fffffffu) : ~u);
}

__device__ __forceinline__ float lrelu(float e) {
  return e > 0.f ? e : NEG_SLOPE * e;
}

// bf16 round-to-nearest-even helpers (bit level, for the W pre-pack).
__device__ __forceinline__ unsigned f2bf(float f) {
  unsigned u = __float_as_uint(f);
  return (u + 0x7fffu + ((u >> 16) & 1u)) >> 16;
}
__device__ __forceinline__ float bf2f(unsigned b) {
  return __uint_as_float(b << 16);
}

// -------------------------------------------------------------------------
// Kernel 0: repack W[128][128] (row-major f32) into split-bf16 B fragments.
// Fragment index t = ((col_tile*4 + kb)*32 + lane); each t owns 8 dwords
// (16 bf16, packed pairs along K) for hi and lo. B layout (32x16, 16-bit):
// lane half selects K 0-7,16-23 / 8-15,24-31; pair v holds K = kb*32 + bkoff + 2v (+8 for v>=4).
// -------------------------------------------------------------------------
__global__ void gat_pack_w(const float* __restrict__ W,
                           unsigned* __restrict__ bhi,
                           unsigned* __restrict__ blo) {
  const int t = blockIdx.x * blockDim.x + threadIdx.x;
  if (t >= 8 * 4 * 32) return;
  const int lane = t & 31;
  const int kb   = (t >> 5) & 3;
  const int ct   = t >> 7;
  const int bcol  = ct * 16 + (lane & 15);
  const int bkoff = (lane >> 4) * 8;
  unsigned* ph = bhi + (size_t)t * 8;
  unsigned* pl = blo + (size_t)t * 8;
#pragma unroll
  for (int v = 0; v < 8; ++v) {
    const int k = kb * 32 + bkoff + 2 * v + (v >= 4 ? 8 : 0);
    const float f0 = W[(size_t)k * F_DIM + bcol];
    const float f1 = W[(size_t)(k + 1) * F_DIM + bcol];
    const unsigned h0 = f2bf(f0), h1 = f2bf(f1);
    const unsigned l0 = f2bf(f0 - bf2f(h0)), l1 = f2bf(f1 - bf2f(h1));
    ph[v] = h0 | (h1 << 16);
    pl[v] = l0 | (l1 << 16);
  }
}

// -------------------------------------------------------------------------
// Kernel 1: h = x @ W, one 16x16 output tile per wave, K=128 fully unrolled
// (4 k-steps x 3 split-precision WMMAs = 12 v_wmma per tile).
// -------------------------------------------------------------------------
__global__ void __launch_bounds__(128)
gat_gemm_wmma(const float* __restrict__ x,
              const unsigned* __restrict__ bhi_frag,
              const unsigned* __restrict__ blo_frag,
              float* __restrict__ h, int N, int n_tiles) {
  const int lane = threadIdx.x & 31;
  const int wave = threadIdx.x >> 5;
  const int tile = blockIdx.x * 4 + wave;
  if (tile >= n_tiles) return;

  const int row_tile = tile >> 3;        // 8 column tiles of 16 (F_DIM = 128)
  const int col_tile = tile & 7;
  const int m0 = row_tile * 16;
  const int n0 = col_tile * 16;

  const int half = lane >> 4;            // 0: lanes 0-15, 1: lanes 16-31
  const int lmod = lane & 15;
  int arow = m0 + lmod;                  // A-fragment row for this lane
  if (arow >= N) arow = N - 1;           // clamp (padding rows never stored)
  const int akoff = half * 8;            // A: K offset within 16-wide half

  const float* xr0 = x + (size_t)arow * F_DIM;

  v8f acc = {};
#pragma unroll
  for (int kb = 0; kb < 4; ++kb) {
    const float* xr = xr0 + kb * 32;
    // A layout (16-bit A 16x32): pairs v=0..3 -> K = akoff+2v; v=4..7 -> +16.
    // Per lane that is two aligned contiguous 8-float runs -> 4x float4.
    const float4 a0 = *(const float4*)(xr + akoff);
    const float4 a1 = *(const float4*)(xr + akoff + 4);
    const float4 a2 = *(const float4*)(xr + 16 + akoff);
    const float4 a3 = *(const float4*)(xr + 16 + akoff + 4);
    float af[16] = {a0.x, a0.y, a0.z, a0.w, a1.x, a1.y, a1.z, a1.w,
                    a2.x, a2.y, a2.z, a2.w, a3.x, a3.y, a3.z, a3.w};
    v16bf ahi, alo;
#pragma unroll
    for (int i = 0; i < 16; ++i) {
      const __bf16 hi = (__bf16)af[i];
      ahi[i] = hi;
      alo[i] = (__bf16)(af[i] - (float)hi);
    }
    // Pre-packed B fragments: one contiguous 32B load each.
    const size_t fidx = ((size_t)(col_tile * 4 + kb) * 32 + lane) * 8;
    const v16bf bhi = __builtin_bit_cast(v16bf, *(const u8v*)(bhi_frag + fidx));
    const v16bf blo = __builtin_bit_cast(v16bf, *(const u8v*)(blo_frag + fidx));

    acc = __builtin_amdgcn_wmma_f32_16x16x32_bf16(false, ahi, false, bhi,
                                                  (short)0, acc, false, false);
    acc = __builtin_amdgcn_wmma_f32_16x16x32_bf16(false, ahi, false, blo,
                                                  (short)0, acc, false, false);
    acc = __builtin_amdgcn_wmma_f32_16x16x32_bf16(false, alo, false, bhi,
                                                  (short)0, acc, false, false);
  }

  // C/D layout: lane half selects M base (0 or 8), VGPR r -> row m0 + r + 8*half.
  const int bcol = n0 + lmod;
  float* hp = h + (size_t)(m0 + 8 * half) * F_DIM + bcol;
  if (m0 + 16 <= N) {
#pragma unroll
    for (int r = 0; r < 8; ++r) hp[(size_t)r * F_DIM] = acc[r];
  } else {
#pragma unroll
    for (int r = 0; r < 8; ++r)
      if (m0 + 8 * half + r < N) hp[(size_t)r * F_DIM] = acc[r];
  }
}

// -------------------------------------------------------------------------
// Kernel 2: per (node, head): a_src/a_dst dots; init mord=-inf, ssum=0, out=bias.
// -------------------------------------------------------------------------
__global__ void gat_prep(const float* __restrict__ h,
                         const float* __restrict__ att_src,
                         const float* __restrict__ att_dst,
                         const float* __restrict__ bias,
                         float* __restrict__ a_src, float* __restrict__ a_dst,
                         unsigned* __restrict__ mord, float* __restrict__ ssum,
                         float* __restrict__ out, int N) {
  const int t = blockIdx.x * blockDim.x + threadIdx.x;
  if (t >= N * H_HEADS) return;
  const int n  = t >> 2;
  const int hh = t & 3;
  const float* hp = h + (size_t)n * F_DIM + hh * C_CH;
  const float* as = att_src + hh * C_CH;
  const float* ad = att_dst + hh * C_CH;
  const float* bp = bias + hh * C_CH;
  float* op = out + (size_t)n * F_DIM + hh * C_CH;
  float s = 0.f, d = 0.f;
#pragma unroll
  for (int c = 0; c < C_CH; ++c) {
    float v = hp[c];
    s += v * as[c];
    d += v * ad[c];
    op[c] = bp[c];
  }
  a_src[t] = s;
  a_dst[t] = d;
  mord[t]  = f2ord(-INFINITY);
  ssum[t]  = 0.f;
}

__device__ __forceinline__ void edge_nodes(const int* __restrict__ ei, int E,
                                           int i, int& s, int& d) {
  if (i < E) { s = ei[i]; d = ei[E + i]; }
  else       { s = i - E; d = i - E; }    // self loops appended
}

// -------------------------------------------------------------------------
// Kernel 3: segment max of leaky-relu logits (ordered-uint atomicMax).
// -------------------------------------------------------------------------
__global__ void gat_edge_max(const int* __restrict__ ei,
                             const float* __restrict__ a_src,
                             const float* __restrict__ a_dst,
                             unsigned* __restrict__ mord, int E, int N) {
  const int i = blockIdx.x * blockDim.x + threadIdx.x;
  if (i >= E + N) return;
  int s, d;
  edge_nodes(ei, E, i, s, d);
  const float4 av = *(const float4*)(a_src + (size_t)s * 4);
  const float4 bv = *(const float4*)(a_dst + (size_t)d * 4);
  float e[4] = {av.x + bv.x, av.y + bv.y, av.z + bv.z, av.w + bv.w};
#pragma unroll
  for (int hh = 0; hh < 4; ++hh)
    atomicMax(&mord[(size_t)d * 4 + hh], f2ord(lrelu(e[hh])));
}

// -------------------------------------------------------------------------
// Kernel 4: segment sum of exp(e - m).
// -------------------------------------------------------------------------
__global__ void gat_edge_sum(const int* __restrict__ ei,
                             const float* __restrict__ a_src,
                             const float* __restrict__ a_dst,
                             const unsigned* __restrict__ mord,
                             float* __restrict__ ssum, int E, int N) {
  const int i = blockIdx.x * blockDim.x + threadIdx.x;
  if (i >= E + N) return;
  int s, d;
  edge_nodes(ei, E, i, s, d);
  const float4 av = *(const float4*)(a_src + (size_t)s * 4);
  const float4 bv = *(const float4*)(a_dst + (size_t)d * 4);
  float e[4] = {av.x + bv.x, av.y + bv.y, av.z + bv.z, av.w + bv.w};
#pragma unroll
  for (int hh = 0; hh < 4; ++hh) {
    float m = ord2f(mord[(size_t)d * 4 + hh]);
    atomicAdd(&ssum[(size_t)d * 4 + hh], __expf(lrelu(e[hh]) - m));
  }
}

// -------------------------------------------------------------------------
// Kernel 5: one wave32 per edge; out[dst] += alpha * h[src] over 128 channels.
// Iteration i (32 lanes) == head i, so alpha[i] is wave-uniform per iteration.
// -------------------------------------------------------------------------
__global__ void __launch_bounds__(256)
gat_edge_scatter(const int* __restrict__ ei,
                 const float* __restrict__ a_src,
                 const float* __restrict__ a_dst,
                 const unsigned* __restrict__ mord,
                 const float* __restrict__ ssum,
                 const float* __restrict__ h,
                 float* __restrict__ out, int E, int N) {
  const int lane = threadIdx.x & 31;
  const int wid  = (blockIdx.x * blockDim.x + threadIdx.x) >> 5;
  if (wid >= E + N) return;
  int s, d;
  edge_nodes(ei, E, wid, s, d);
  const float4 av = *(const float4*)(a_src + (size_t)s * 4);
  const float4 bv = *(const float4*)(a_dst + (size_t)d * 4);
  const float4 sv = *(const float4*)(ssum + (size_t)d * 4);
  float e[4]  = {av.x + bv.x, av.y + bv.y, av.z + bv.z, av.w + bv.w};
  float ss[4] = {sv.x, sv.y, sv.z, sv.w};
  float alpha[4];
#pragma unroll
  for (int hh = 0; hh < 4; ++hh) {
    float m = ord2f(mord[(size_t)d * 4 + hh]);
    alpha[hh] = __expf(lrelu(e[hh]) - m) / (ss[hh] + EPS_F);
  }
  const float* hs = h + (size_t)s * F_DIM;
  float* od = out + (size_t)d * F_DIM;
#pragma unroll
  for (int i = 0; i < 4; ++i) {
    const int c = i * 32 + lane;
    atomicAdd(&od[c], alpha[i] * hs[c]);
  }
}

// -------------------------------------------------------------------------
extern "C" void kernel_launch(void* const* d_in, const int* in_sizes, int n_in,
                              void* d_out, int out_size, void* d_ws, size_t ws_size,
                              hipStream_t stream) {
  const float* x       = (const float*)d_in[0];
  const int*   ei      = (const int*)d_in[1];
  const float* W       = (const float*)d_in[2];
  const float* att_src = (const float*)d_in[3];
  const float* att_dst = (const float*)d_in[4];
  const float* bias    = (const float*)d_in[5];
  float* out = (float*)d_out;

  const int N = in_sizes[0] / F_DIM;
  const int E = in_sizes[1] / 2;
  const int Etot = E + N;

  // Workspace carve-up.
  char* ws = (char*)d_ws;
  float* h = (float*)ws;           ws += (size_t)N * F_DIM * sizeof(float);
  float* a_src = (float*)ws;       ws += (size_t)N * H_HEADS * sizeof(float);
  float* a_dst = (float*)ws;       ws += (size_t)N * H_HEADS * sizeof(float);
  unsigned* mord = (unsigned*)ws;  ws += (size_t)N * H_HEADS * sizeof(unsigned);
  float* ssum = (float*)ws;        ws += (size_t)N * H_HEADS * sizeof(float);
  unsigned* bhi = (unsigned*)ws;   ws += (size_t)8 * 4 * 32 * 8 * sizeof(unsigned);
  unsigned* blo = (unsigned*)ws;

  // 0) repack W into split-bf16 WMMA B fragments (1024 work items, one shot)
  gat_pack_w<<<4, 256, 0, stream>>>(W, bhi, blo);

  // 1) GEMM: 16x16 tile per wave, 4 waves per 128-thread block
  const int row_tiles = (N + 15) / 16;
  const int n_tiles = row_tiles * (F_DIM / 16);
  gat_gemm_wmma<<<(n_tiles + 3) / 4, 128, 0, stream>>>(x, bhi, blo, h, N, n_tiles);

  // 2) prep
  const int nt2 = N * H_HEADS;
  gat_prep<<<(nt2 + 255) / 256, 256, 0, stream>>>(h, att_src, att_dst, bias,
                                                  a_src, a_dst, mord, ssum, out, N);

  // 3) segment max
  gat_edge_max<<<(Etot + 255) / 256, 256, 0, stream>>>(ei, a_src, a_dst, mord, E, N);

  // 4) segment sum
  gat_edge_sum<<<(Etot + 255) / 256, 256, 0, stream>>>(ei, a_src, a_dst, mord, ssum, E, N);

  // 5) scatter: 8 waves (edges) per 256-thread block
  gat_edge_scatter<<<(Etot + 7) / 8, 256, 0, stream>>>(ei, a_src, a_dst, mord, ssum,
                                                       h, out, E, N);
}
